// ConformerBlock_76304388981439
// MI455X (gfx1250) — matrix operand order repaired and checked by reading, not compiled, not run on hardware
//
#include <hip/hip_runtime.h>

// ---------------------------------------------------------------------------
// Conformer block for MI455X (gfx1250), wave32 + WMMA f32_16x16x32_f16.
// B=16, N=1024, DIM=512, H=8, DH=64, FF=2048, CIN=1024, K=31, MAXPOS=512
// ---------------------------------------------------------------------------

#define TOKS   16384   // B*N
#define DIMC   512
#define FFC    2048
#define HEADS  8
#define DHC    64
#define SEQ    1024
#define CINC   1024
#define KWIN   31
#define MAXP   512

typedef __attribute__((ext_vector_type(16))) _Float16 v16h;
typedef __attribute__((ext_vector_type(8)))  _Float16 v8h;
typedef __attribute__((ext_vector_type(8)))  float    v8f;

#define CDIV(a, b) (((a) + (b) - 1) / (b))

__device__ __forceinline__ float sigf(float x) { return 1.0f / (1.0f + __expf(-x)); }

__device__ __forceinline__ v8f wmma_f16(v16h a, v16h b, v8f c) {
  return __builtin_amdgcn_wmma_f32_16x16x32_f16(false, a, false, b, (short)0, c,
                                                false, false);
}

// A fragment (16x32, M x K), per ISA 7.12.2:
// lanes 0-15 (M=lane):  halves 0..7 = K 0..7,  halves 8..15 = K 16..23
// lanes 16-31(M=ln-16): halves 0..7 = K 8..15, halves 8..15 = K 24..31
__device__ __forceinline__ v16h load_a_frag(const _Float16* t, int ld, int row0,
                                            int k0, int lane) {
  const int m = lane & 15, hi = lane >> 4;
  const _Float16* p = t + (size_t)(row0 + m) * ld + k0 + hi * 8;
  v16h f;
#pragma unroll
  for (int i = 0; i < 8; ++i) { f[i] = p[i]; f[i + 8] = p[i + 16]; }
  return f;
}

// B fragment (32x16, K x N) from BT stored row-major [N][K]:
// lanes 0-15: col n, K 0..7,16..23; lanes 16-31: col n, K 8..15,24..31
__device__ __forceinline__ v16h load_b_frag(const _Float16* t, int ld, int col0,
                                            int k0, int lane) {
  const int n = lane & 15, khi = lane >> 4;
  const _Float16* p = t + (size_t)(col0 + n) * ld + k0 + khi * 8;
  v16h f;
#pragma unroll
  for (int i = 0; i < 8; ++i) { f[i] = p[i]; f[i + 8] = p[i + 16]; }
  return f;
}

// ---------------------------------------------------------------------------
// Weight preparation: f32 [K][N] -> f16 BT [N][K]; plain f16 convert.
// ---------------------------------------------------------------------------
__global__ __launch_bounds__(256) void w_transpose_f16(
    const float* __restrict__ src, _Float16* __restrict__ dst, int K, int N) {
  size_t i = (size_t)blockIdx.x * 256 + threadIdx.x;
  size_t total = (size_t)K * N;
  if (i >= total) return;
  int n = (int)(i / K), k = (int)(i % K);
  dst[i] = (_Float16)src[(size_t)k * N + n];
}

__global__ __launch_bounds__(256) void w_convert_f16(
    const float* __restrict__ src, _Float16* __restrict__ dst, int total) {
  int i = blockIdx.x * 256 + threadIdx.x;
  if (i < total) dst[i] = (_Float16)src[i];
}

// ---------------------------------------------------------------------------
// LayerNorm over 512 cols, one row per 256-thread block.
// ---------------------------------------------------------------------------
template <bool OUT16>
__global__ __launch_bounds__(256) void ln_kernel(
    const float* __restrict__ x, const float* __restrict__ g,
    const float* __restrict__ bt, float* __restrict__ o32,
    _Float16* __restrict__ o16) {
  const int row = blockIdx.x, tid = threadIdx.x;
  const float* xr = x + (size_t)row * DIMC;
  float v0 = xr[tid], v1 = xr[tid + 256];
  float s = v0 + v1, q = v0 * v0 + v1 * v1;
#pragma unroll
  for (int off = 16; off >= 1; off >>= 1) {
    s += __shfl_xor(s, off, 32);
    q += __shfl_xor(q, off, 32);
  }
  __shared__ float rs[8], rq[8];
  const int lane = tid & 31, wid = tid >> 5;
  if (lane == 0) { rs[wid] = s; rq[wid] = q; }
  __syncthreads();
  float ts = 0.f, tq = 0.f;
#pragma unroll
  for (int i = 0; i < 8; ++i) { ts += rs[i]; tq += rq[i]; }
  const float mu = ts * (1.0f / DIMC);
  const float var = tq * (1.0f / DIMC) - mu * mu;
  const float r = rsqrtf(var + 1e-5f);
  const float y0 = (v0 - mu) * r * g[tid] + bt[tid];
  const float y1 = (v1 - mu) * r * g[tid + 256] + bt[tid + 256];
  const size_t base = (size_t)row * DIMC;
  if (OUT16) {
    o16[base + tid] = (_Float16)y0;
    o16[base + tid + 256] = (_Float16)y1;
  } else {
    o32[base + tid] = y0;
    o32[base + tid + 256] = y1;
  }
}

// ---------------------------------------------------------------------------
// WMMA GEMM: C[M,N] = A[M,K](f16) x BT[N,K](f16) + epilogue.
// Block tile 128x64, 8 waves (4x2), wave tile 32x32 = 2x2 WMMA frags, KTILE 64.
// ---------------------------------------------------------------------------
enum { EPI_BIAS16 = 0, EPI_SWISH16 = 1, EPI_HALFRES32 = 2, EPI_RES32 = 3 };

template <int EPI>
__global__ __launch_bounds__(256) void gemm_wmma_kernel(
    const _Float16* __restrict__ A, const _Float16* __restrict__ BT,
    const float* __restrict__ bias, const float* __restrict__ resid,
    float* __restrict__ out32, _Float16* __restrict__ out16, int N, int K) {
  __shared__ __align__(16) _Float16 As[128 * 72];
  __shared__ __align__(16) _Float16 Bs[64 * 72];
  const int tid = threadIdx.x;
  const int lane = tid & 31, wid = tid >> 5;
  const int wm = wid >> 1, wn = wid & 1;
  const int row0 = blockIdx.y * 128;
  const int col0 = blockIdx.x * 64;

  v8f acc[2][2] = {};

  for (int k0 = 0; k0 < K; k0 += 64) {
    __syncthreads();
#pragma unroll
    for (int t = 0; t < 4; ++t) {
      int c = tid + t * 256;
      int r = c >> 3, kc = (c & 7) * 8;
      *(v8h*)&As[r * 72 + kc] = *(const v8h*)&A[(size_t)(row0 + r) * K + k0 + kc];
    }
#pragma unroll
    for (int t = 0; t < 2; ++t) {
      int c = tid + t * 256;
      int r = c >> 3, kc = (c & 7) * 8;
      *(v8h*)&Bs[r * 72 + kc] = *(const v8h*)&BT[(size_t)(col0 + r) * K + k0 + kc];
    }
    if (k0 + 64 < K)  // next A tile into GL2 (global_prefetch_b8)
      __builtin_prefetch(&A[(size_t)(row0 + (tid >> 1)) * K + k0 + 64], 0, 1);
    __syncthreads();
#pragma unroll
    for (int ks = 0; ks < 2; ++ks) {
      v16h a0 = load_a_frag(As, 72, wm * 32, ks * 32, lane);
      v16h a1 = load_a_frag(As, 72, wm * 32 + 16, ks * 32, lane);
      v16h b0 = load_b_frag(Bs, 72, wn * 32, ks * 32, lane);
      v16h b1 = load_b_frag(Bs, 72, wn * 32 + 16, ks * 32, lane);
      acc[0][0] = wmma_f16(a0, b0, acc[0][0]);
      acc[0][1] = wmma_f16(a0, b1, acc[0][1]);
      acc[1][0] = wmma_f16(a1, b0, acc[1][0]);
      acc[1][1] = wmma_f16(a1, b1, acc[1][1]);
    }
  }

  // epilogue: C element (m,n): vgpr r, lane = n16 + 16*(m>=8)
  const int hi = lane >> 4, nl = lane & 15;
#pragma unroll
  for (int mi = 0; mi < 2; ++mi) {
#pragma unroll
    for (int ni = 0; ni < 2; ++ni) {
      const int col = col0 + wn * 32 + ni * 16 + nl;
      const float bv = bias ? bias[col] : 0.0f;
#pragma unroll
      for (int r = 0; r < 8; ++r) {
        const int row = row0 + wm * 32 + mi * 16 + r + 8 * hi;
        const size_t o = (size_t)row * N + col;
        float v = acc[mi][ni][r] + bv;
        if (EPI == EPI_SWISH16) {
          v = v * sigf(v);
          out16[o] = (_Float16)v;
        } else if (EPI == EPI_BIAS16) {
          out16[o] = (_Float16)v;
        } else if (EPI == EPI_HALFRES32) {
          out32[o] = 0.5f * v + resid[o];
        } else {
          out32[o] = v + resid[o];
        }
      }
    }
  }
}

// ---------------------------------------------------------------------------
// Attention: one WG per (b, h, 64-row q tile); 4 waves, 16 q rows each.
// Flash softmax over 16 j-tiles of 64. Relative positions via WMMA against a
// 128-row rel_emb window: c = (i-i0)-(j-j0)+63, rel row = clamp(d0+c,0,1024)
// which equals clip(i-j,-512,512)+512 exactly.
// ---------------------------------------------------------------------------
__global__ __launch_bounds__(128) void attn_kernel(
    const _Float16* __restrict__ q16, const _Float16* __restrict__ kv16,
    const _Float16* __restrict__ rel16, _Float16* __restrict__ ctx16) {
  __shared__ __align__(16) _Float16 smem[31232];  // 62.5 KB
  _Float16* Qs  = smem;          // 64x72
  _Float16* KPs = smem + 4608;   // 64x72 : K tile, later aliased as P tile
  _Float16* Vts = smem + 9216;   // 64x72 : V transposed [d][j]
  _Float16* Rs  = smem + 13824;  // 128x72: rel window (BT layout)
  _Float16* PRs = smem + 23040;  // 4 waves x 16 x 128 : q.rel scores

  const int tid = threadIdx.x, lane = tid & 31, w = tid >> 5;
  const int hi = lane >> 4, nl = lane & 15;
  const int gid = blockIdx.x;
  const int it = gid & 15, h = (gid >> 4) & 7, b = gid >> 7;
  const int i0 = it * 64;
  const size_t rowb = (size_t)b * SEQ;
  const float scale = 0.125f;  // DH^-0.5

  // load Q tile (64 x 64)
#pragma unroll
  for (int t = 0; t < 4; ++t) {
    int c = tid + t * 128;
    int r = c >> 3, kc = (c & 7) * 8;
    *(v8h*)&Qs[r * 72 + kc] =
        *(const v8h*)&q16[(rowb + i0 + r) * 512 + h * 64 + kc];
  }

  v8f Ov[4] = {};
  float l[8], mx[8];
#pragma unroll
  for (int r = 0; r < 8; ++r) { l[r] = 0.0f; mx[r] = -1e30f; }

  for (int jt = 0; jt < 16; ++jt) {
    const int j0 = jt * 64;
    __syncthreads();  // previous iteration fully consumed KPs/Vts/Rs
    // K tile (64 x 64)
#pragma unroll
    for (int t = 0; t < 4; ++t) {
      int c = tid + t * 128;
      int r = c >> 3, kc = (c & 7) * 8;
      *(v8h*)&KPs[r * 72 + kc] =
          *(const v8h*)&kv16[(rowb + j0 + r) * 1024 + h * 64 + kc];
    }
    // V tile transposed: Vts[d][j]
#pragma unroll
    for (int t = 0; t < 4; ++t) {
      int c = tid + t * 128;
      int j = c >> 3, dc = (c & 7) * 8;
      v8h vv = *(const v8h*)&kv16[(rowb + j0 + j) * 1024 + 512 + h * 64 + dc];
#pragma unroll
      for (int e = 0; e < 8; ++e) Vts[(dc + e) * 72 + j] = vv[e];
    }
    // rel window rows d0..d0+127 (clamped)
    const int d0 = i0 - j0 - 63 + MAXP;
#pragma unroll
    for (int t = 0; t < 8; ++t) {
      int c = tid + t * 128;
      int cr = c >> 3, kc = (c & 7) * 8;
      int d = d0 + cr;
      d = d < 0 ? 0 : (d > 2 * MAXP ? 2 * MAXP : d);
      *(v8h*)&Rs[cr * 72 + kc] = *(const v8h*)&rel16[(size_t)d * 64 + kc];
    }
    __syncthreads();

    // S = Q K^T (16x64 strip per wave)
    v16h aq0 = load_a_frag(Qs, 72, w * 16, 0, lane);
    v16h aq1 = load_a_frag(Qs, 72, w * 16, 32, lane);
    v8f sacc[4] = {};
#pragma unroll
    for (int nt = 0; nt < 4; ++nt) {
      sacc[nt] = wmma_f16(aq0, load_b_frag(KPs, 72, nt * 16, 0, lane), sacc[nt]);
      sacc[nt] = wmma_f16(aq1, load_b_frag(KPs, 72, nt * 16, 32, lane), sacc[nt]);
    }
    // PR strip: 16 x 128 of q.rel, one ctile at a time -> LDS
#pragma unroll
    for (int ct = 0; ct < 8; ++ct) {
      v8f pr = {};
      pr = wmma_f16(aq0, load_b_frag(Rs, 72, ct * 16, 0, lane), pr);
      pr = wmma_f16(aq1, load_b_frag(Rs, 72, ct * 16, 32, lane), pr);
#pragma unroll
      for (int r = 0; r < 8; ++r)
        PRs[w * 2048 + (r + 8 * hi) * 128 + ct * 16 + nl] = (_Float16)pr[r];
    }
    __syncthreads();  // all waves done with KPs(K)/Rs; safe to alias KPs as P

    // streaming softmax per strip row
    float sv[4][8];
#pragma unroll
    for (int r = 0; r < 8; ++r) {
      const int m = r + 8 * hi;
      float rm = -1e30f;
#pragma unroll
      for (int nt = 0; nt < 4; ++nt) {
        const int jl = nt * 16 + nl;
        const int cc = w * 16 + m - jl + 63;  // 0..126
        const float s =
            (sacc[nt][r] + (float)PRs[w * 2048 + m * 128 + cc]) * scale;
        sv[nt][r] = s;
        rm = fmaxf(rm, s);
      }
#pragma unroll
      for (int off = 8; off >= 1; off >>= 1) rm = fmaxf(rm, __shfl_xor(rm, off, 32));
      const float newm = fmaxf(mx[r], rm);
      const float alpha = __expf(mx[r] - newm);
      float ps = 0.0f;
#pragma unroll
      for (int nt = 0; nt < 4; ++nt) {
        const float p = __expf(sv[nt][r] - newm);
        sv[nt][r] = p;
        ps += p;
      }
#pragma unroll
      for (int off = 8; off >= 1; off >>= 1) ps += __shfl_xor(ps, off, 32);
      l[r] = l[r] * alpha + ps;
#pragma unroll
      for (int dt = 0; dt < 4; ++dt) Ov[dt][r] *= alpha;
      mx[r] = newm;
#pragma unroll
      for (int nt = 0; nt < 4; ++nt)  // P -> LDS (A layout reload), own strip
        KPs[(w * 16 + m) * 72 + nt * 16 + nl] = (_Float16)sv[nt][r];
    }

    // O += P V  (same-wave LDS ops are in-order; no cross-wave use of strip)
    v16h ap0 = load_a_frag(KPs, 72, w * 16, 0, lane);
    v16h ap1 = load_a_frag(KPs, 72, w * 16, 32, lane);
#pragma unroll
    for (int dt = 0; dt < 4; ++dt) {
      Ov[dt] = wmma_f16(ap0, load_b_frag(Vts, 72, dt * 16, 0, lane), Ov[dt]);
      Ov[dt] = wmma_f16(ap1, load_b_frag(Vts, 72, dt * 16, 32, lane), Ov[dt]);
    }
  }

  // normalize + write ctx (f16 [T, 512])
#pragma unroll
  for (int dt = 0; dt < 4; ++dt) {
#pragma unroll
    for (int r = 0; r < 8; ++r) {
      const int m = r + 8 * hi;
      const size_t row = rowb + i0 + w * 16 + m;
      const int col = h * 64 + dt * 16 + nl;
      ctx16[row * 512 + col] = (_Float16)(Ov[dt][r] / l[r]);
    }
  }
}

// ---------------------------------------------------------------------------
// GLU: a * sigmoid(g) over conv1 output [T, 2048] -> [T, 1024]
// ---------------------------------------------------------------------------
__global__ __launch_bounds__(256) void glu_kernel(const _Float16* __restrict__ in,
                                                  _Float16* __restrict__ out) {
  size_t idx = (size_t)blockIdx.x * 256 + threadIdx.x;  // < TOKS*CINC
  size_t t = idx >> 10;
  int c = (int)(idx & 1023);
  float a = (float)in[t * 2048 + c];
  float g = (float)in[t * 2048 + 1024 + c];
  out[idx] = (_Float16)(a * sigf(g));
}

// ---------------------------------------------------------------------------
// Causal depthwise conv K=31 (+bias +swish), channels-last f16.
// ---------------------------------------------------------------------------
__global__ __launch_bounds__(256) void dwconv_kernel(
    const _Float16* __restrict__ in, const float* __restrict__ wgt,
    const float* __restrict__ bias, _Float16* __restrict__ out) {
  size_t idx = (size_t)blockIdx.x * 256 + threadIdx.x;  // < TOKS*CINC
  int c = (int)(idx & 1023);
  size_t t = idx >> 10;
  int n = (int)(t & 1023);
  size_t bb = t >> 10;
  float acc = bias[c];
#pragma unroll
  for (int k = 0; k < KWIN; ++k) {
    int nn = n - (KWIN - 1) + k;
    if (nn >= 0)
      acc += (float)in[((bb << 10) + nn) * 1024 + c] * wgt[c * KWIN + k];
  }
  out[idx] = (_Float16)(acc * sigf(acc));
}

// ---------------------------------------------------------------------------
// host-side orchestration
// ---------------------------------------------------------------------------
extern "C" void kernel_launch(void* const* d_in, const int* in_sizes, int n_in,
                              void* d_out, int out_size, void* d_ws,
                              size_t ws_size, hipStream_t stream) {
  (void)in_sizes; (void)n_in; (void)out_size; (void)ws_size;
  const float* x        = (const float*)d_in[0];
  const float* ff1_ln_g = (const float*)d_in[1];
  const float* ff1_ln_b = (const float*)d_in[2];
  const float* ff1_w1   = (const float*)d_in[3];
  const float* ff1_b1   = (const float*)d_in[4];
  const float* ff1_w2   = (const float*)d_in[5];
  const float* ff1_b2   = (const float*)d_in[6];
  const float* attn_ln_g= (const float*)d_in[7];
  const float* attn_ln_b= (const float*)d_in[8];
  const float* wq       = (const float*)d_in[9];
  const float* wkv      = (const float*)d_in[10];
  const float* wo       = (const float*)d_in[11];
  const float* bo       = (const float*)d_in[12];
  const float* rel_emb  = (const float*)d_in[13];
  const float* conv_ln_g= (const float*)d_in[14];
  const float* conv_ln_b= (const float*)d_in[15];
  const float* conv1_w  = (const float*)d_in[16];
  const float* conv1_b  = (const float*)d_in[17];
  const float* dw_w     = (const float*)d_in[18];
  const float* dw_b     = (const float*)d_in[19];
  const float* conv2_w  = (const float*)d_in[20];
  const float* conv2_b  = (const float*)d_in[21];
  const float* ff2_ln_g = (const float*)d_in[22];
  const float* ff2_ln_b = (const float*)d_in[23];
  const float* ff2_w1   = (const float*)d_in[24];
  const float* ff2_b1   = (const float*)d_in[25];
  const float* ff2_w2   = (const float*)d_in[26];
  const float* ff2_b2   = (const float*)d_in[27];
  const float* post_ln_g= (const float*)d_in[28];
  const float* post_ln_b= (const float*)d_in[29];

  char* ws = (char*)d_ws;
  float*     resid = (float*)(ws + 0);                   // f32 [T,512]
  _Float16*  scr16 = (_Float16*)(ws + 33554432);         // f16 [T,2048]
  _Float16*  ln16  = (_Float16*)(ws + 100663296);        // f16 [T,512]
  _Float16*  q16   = (_Float16*)(ws + 117440512);        // f16 [T,512]
  _Float16*  kv16  = (_Float16*)(ws + 134217728);        // f16 [T,1024]
  _Float16*  ctx16 = (_Float16*)(ws + 167772160);        // f16 [T,512]
  _Float16*  glu16 = (_Float16*)(ws + 117440512);        // alias q16+kv16(lo), post-attn
  _Float16*  dw16  = (_Float16*)(ws + 150994944);        // alias kv16(hi)+ctx16, post-attn
  _Float16*  wf    = (_Float16*)(ws + 184549376);        // weight arena (f16)

  _Float16* ff1w1T = wf;                                  // [2048][512]
  _Float16* ff1w2T = ff1w1T + 2048 * 512;                 // [512][2048]
  _Float16* wqT    = ff1w2T + 512 * 2048;                 // [512][512]
  _Float16* wkvT   = wqT + 512 * 512;                     // [1024][512]
  _Float16* woT    = wkvT + 1024 * 512;                   // [512][512]
  _Float16* c1T    = woT + 512 * 512;                     // [2048][512]
  _Float16* c2T    = c1T + 2048 * 512;                    // [512][1024]
  _Float16* ff2w1T = c2T + 512 * 1024;                    // [2048][512]
  _Float16* ff2w2T = ff2w1T + 2048 * 512;                 // [512][2048]
  _Float16* rel16  = ff2w2T + 512 * 2048;                 // [1025][64]

  // --- weight conversion ---
  w_transpose_f16<<<CDIV(512 * 2048, 256), 256, 0, stream>>>(ff1_w1, ff1w1T, 512, 2048);
  w_transpose_f16<<<CDIV(2048 * 512, 256), 256, 0, stream>>>(ff1_w2, ff1w2T, 2048, 512);
  w_transpose_f16<<<CDIV(512 * 512, 256), 256, 0, stream>>>(wq, wqT, 512, 512);
  w_transpose_f16<<<CDIV(512 * 1024, 256), 256, 0, stream>>>(wkv, wkvT, 512, 1024);
  w_transpose_f16<<<CDIV(512 * 512, 256), 256, 0, stream>>>(wo, woT, 512, 512);
  w_transpose_f16<<<CDIV(512 * 2048, 256), 256, 0, stream>>>(conv1_w, c1T, 512, 2048);
  w_transpose_f16<<<CDIV(1024 * 512, 256), 256, 0, stream>>>(conv2_w, c2T, 1024, 512);
  w_transpose_f16<<<CDIV(512 * 2048, 256), 256, 0, stream>>>(ff2_w1, ff2w1T, 512, 2048);
  w_transpose_f16<<<CDIV(2048 * 512, 256), 256, 0, stream>>>(ff2_w2, ff2w2T, 2048, 512);
  w_convert_f16<<<CDIV(1025 * 64, 256), 256, 0, stream>>>(rel_emb, rel16, 1025 * 64);

  // residual stream = x
  hipMemcpyAsync(resid, x, (size_t)TOKS * DIMC * sizeof(float),
                 hipMemcpyDeviceToDevice, stream);

  const dim3 blk(256);
  const dim3 gFF(FFC / 64, TOKS / 128);
  const dim3 gD (DIMC / 64, TOKS / 128);
  const dim3 gKV(1024 / 64, TOKS / 128);

  // --- FF1: x += 0.5*FF(LN(x)) ---
  ln_kernel<true><<<TOKS, blk, 0, stream>>>(resid, ff1_ln_g, ff1_ln_b, nullptr, ln16);
  gemm_wmma_kernel<EPI_SWISH16><<<gFF, blk, 0, stream>>>(
      ln16, ff1w1T, ff1_b1, nullptr, nullptr, scr16, FFC, DIMC);
  gemm_wmma_kernel<EPI_HALFRES32><<<gD, blk, 0, stream>>>(
      scr16, ff1w2T, ff1_b2, resid, resid, nullptr, DIMC, FFC);

  // --- attention: x += Attn(LN(x)) ---
  ln_kernel<true><<<TOKS, blk, 0, stream>>>(resid, attn_ln_g, attn_ln_b, nullptr, ln16);
  gemm_wmma_kernel<EPI_BIAS16><<<gD, blk, 0, stream>>>(
      ln16, wqT, nullptr, nullptr, nullptr, q16, DIMC, DIMC);
  gemm_wmma_kernel<EPI_BIAS16><<<gKV, blk, 0, stream>>>(
      ln16, wkvT, nullptr, nullptr, nullptr, kv16, 1024, DIMC);
  attn_kernel<<<16 * HEADS * (SEQ / 64), dim3(128), 0, stream>>>(q16, kv16, rel16, ctx16);
  gemm_wmma_kernel<EPI_RES32><<<gD, blk, 0, stream>>>(
      ctx16, woT, bo, resid, resid, nullptr, DIMC, DIMC);

  // --- conv module: x += Conv(x) ---
  ln_kernel<true><<<TOKS, blk, 0, stream>>>(resid, conv_ln_g, conv_ln_b, nullptr, ln16);
  gemm_wmma_kernel<EPI_BIAS16><<<gFF, blk, 0, stream>>>(
      ln16, c1T, conv1_b, nullptr, nullptr, scr16, FFC, DIMC);
  glu_kernel<<<CDIV(TOKS * CINC, 256), blk, 0, stream>>>(scr16, glu16);
  dwconv_kernel<<<CDIV(TOKS * CINC, 256), blk, 0, stream>>>(glu16, dw_w, dw_b, dw16);
  gemm_wmma_kernel<EPI_RES32><<<gD, blk, 0, stream>>>(
      dw16, c2T, conv2_b, resid, resid, nullptr, DIMC, CINC);

  // --- FF2: x += 0.5*FF(LN(x)) ---
  ln_kernel<true><<<TOKS, blk, 0, stream>>>(resid, ff2_ln_g, ff2_ln_b, nullptr, ln16);
  gemm_wmma_kernel<EPI_SWISH16><<<gFF, blk, 0, stream>>>(
      ln16, ff2w1T, ff2_b1, nullptr, nullptr, scr16, FFC, DIMC);
  gemm_wmma_kernel<EPI_HALFRES32><<<gD, blk, 0, stream>>>(
      scr16, ff2w2T, ff2_b2, resid, resid, nullptr, DIMC, FFC);

  // --- post LN -> output (f32) ---
  ln_kernel<false><<<TOKS, blk, 0, stream>>>(resid, post_ln_g, post_ln_b,
                                             (float*)d_out, nullptr);
}
